// Model_80805514707342
// MI455X (gfx1250) — matrix operand order repaired and checked by reading, not compiled, not run on hardware
//
#include <hip/hip_runtime.h>
#include <hip/hip_bf16.h>
#include <stdint.h>

// ---- problem dims ----
#define T_STEPS 4096
#define I_DIM   512
#define H_DIM   2048
#define G_DIM   8192   // 4*H
#define O_DIM   128
#define NB_REC  128    // persistent blocks in the recurrence (2048/16 j-slices)

typedef __attribute__((ext_vector_type(16))) __bf16 v16bf;
typedef __attribute__((ext_vector_type(8)))  float  v8f;
typedef __attribute__((ext_vector_type(4)))  int    v4i;

// address-space-qualified pointer typedefs for the async builtin
typedef __attribute__((address_space(1))) v4i* gptr_v4i;  // global
typedef __attribute__((address_space(3))) v4i* lptr_v4i;  // LDS

union Frag16 { v16bf v; float4 f[2]; };

__device__ __forceinline__ float bf_lo(unsigned u) { return __uint_as_float(u << 16); }
__device__ __forceinline__ float bf_hi(unsigned u) { return __uint_as_float(u & 0xffff0000u); }

// gfx1250 async global->LDS path (ASYNCcnt-tracked), guarded so a builtin
// name/arity mismatch degrades to the plain copy instead of failing compile.
#if defined(__has_builtin)
#if __has_builtin(__builtin_amdgcn_global_load_async_to_lds_b128) && \
    __has_builtin(__builtin_amdgcn_s_wait_asynccnt)
#define USE_ASYNC_LDS 1
#endif
#endif

// ---------------- fp32 -> bf16 conversion ----------------
__global__ void k_cvt_bf16(const float* __restrict__ src,
                           __hip_bfloat16* __restrict__ dst, int n) {
  int i = blockIdx.x * blockDim.x + threadIdx.x;
  if (i < n) dst[i] = __float2bfloat16(src[i]);
}

// ---------------- per-call state init ----------------
__global__ void k_setup(const float* __restrict__ b_ih, const float* __restrict__ b_hh,
                        float* __restrict__ bias, float* __restrict__ hbuf,
                        unsigned* __restrict__ sync) {
  int i = blockIdx.x * blockDim.x + threadIdx.x;
  if (i < G_DIM) bias[i] = b_ih[i] + b_hh[i];
  if (i < 2 * H_DIM) hbuf[i] = 0.f;          // double-buffered h
  if (i == 0) *sync = 0u;
}

// ---------------- GEMM1: gx[T,4H] = Xb[T,I] @ Wb[4H,I]^T + bias ----------------
// block = 256 thr (8 waves); block tile = 16 (M) x 128 (N); wave tile = 16x16
__global__ __launch_bounds__(256) void k_gemm_gx(
    const __hip_bfloat16* __restrict__ Xb,    // [T, I] row-major
    const __hip_bfloat16* __restrict__ Wb,    // [4H, I] row-major
    const float* __restrict__ bias,           // [4H]
    float* __restrict__ gx)                   // [T, 4H]
{
  const int lane = threadIdx.x & 31;
  const int wave = threadIdx.x >> 5;
  const int hlf  = lane >> 4;
  const int l15  = lane & 15;
  const int mbase = blockIdx.y * 16;
  const int nbase = blockIdx.x * 128 + wave * 16;

  const __hip_bfloat16* arow = Xb + (size_t)(mbase + l15) * I_DIM;
  const __hip_bfloat16* brow = Wb + (size_t)(nbase + l15) * I_DIM;

  v8f acc = {0.f, 0.f, 0.f, 0.f, 0.f, 0.f, 0.f, 0.f};
  for (int k = 0; k < I_DIM; k += 32) {
    Frag16 a, b;
    a.f[0] = *(const float4*)(arow + k +      hlf * 8);
    a.f[1] = *(const float4*)(arow + k + 16 + hlf * 8);
    b.f[0] = *(const float4*)(brow + k +      hlf * 8);
    b.f[1] = *(const float4*)(brow + k + 16 + hlf * 8);
    acc = __builtin_amdgcn_wmma_f32_16x16x32_bf16(false, a.v, false, b.v,
                                                  (short)0, acc, false, false);
  }
  const float bv = bias[nbase + l15];
#pragma unroll
  for (int r = 0; r < 8; ++r) {
    int m = mbase + r + hlf * 8;
    gx[(size_t)m * G_DIM + nbase + l15] = acc[r] + bv;
  }
}

// ---------------- recurrence: persistent grid, one grid-sync per timestep ----------------
__global__ __launch_bounds__(256) void k_lstm_steps(
    const __hip_bfloat16* __restrict__ Whh,   // [4H, H] row-major bf16 (L2-resident, 32MB)
    const float* __restrict__ gx,             // [T, 4H]
    float* __restrict__ hbuf,                 // [2, H] double buffer
    __hip_bfloat16* __restrict__ hs,          // [T, H] bf16 (for final GEMM)
    unsigned* __restrict__ sync)
{
  __shared__ float h_lds[H_DIM];
  __shared__ float gsum[4][16];

  const int tid   = threadIdx.x;
  const int lane  = tid & 31;
  const int wave  = tid >> 5;
  const int jbase = blockIdx.x * 16;        // this block owns h[jbase .. jbase+16)
  const int gate  = wave >> 1;              // 0..3 = i,f,g,o
  const int rbase = (wave & 1) * 8;         // row half within the 16-slice

  float c = 0.f;                            // c[jbase+tid] lives in thread tid<16

  for (int t = 0; t < T_STEPS; ++t) {
    const float* hin = hbuf + (t & 1) * H_DIM;
    float*      hout = hbuf + ((t + 1) & 1) * H_DIM;

    // stage h into LDS (2048 f32 = 8KB)
#ifdef USE_ASYNC_LDS
    for (int i = tid; i < H_DIM / 4; i += 256) {
      __builtin_amdgcn_global_load_async_to_lds_b128(
          (gptr_v4i)((const float4*)hin + i),
          (lptr_v4i)((float4*)h_lds + i),
          0, 0);
    }
    __builtin_amdgcn_s_wait_asynccnt(0);
    __syncthreads();
#else
    for (int i = tid; i < H_DIM / 4; i += 256)
      ((float4*)h_lds)[i] = ((const float4*)hin)[i];
    __syncthreads();
#endif

    // each wave: 8 full-length row dots against h (coalesced b128 bf16 loads)
#pragma unroll 1
    for (int r = 0; r < 8; ++r) {
      const int row = gate * H_DIM + jbase + rbase + r;
      const __hip_bfloat16* wrow = Whh + (size_t)row * H_DIM;
      if (r < 7) __builtin_prefetch(wrow + H_DIM + lane * 8, 0, 1);  // global_prefetch_b8
      float sum = 0.f;
#pragma unroll
      for (int i = 0; i < 8; ++i) {
        const int col = lane * 8 + i * 256;
        uint4  wv = *(const uint4*)(wrow + col);
        float4 ha = *(const float4*)(h_lds + col);
        float4 hb = *(const float4*)(h_lds + col + 4);
        sum = fmaf(bf_lo(wv.x), ha.x, sum);
        sum = fmaf(bf_hi(wv.x), ha.y, sum);
        sum = fmaf(bf_lo(wv.y), ha.z, sum);
        sum = fmaf(bf_hi(wv.y), ha.w, sum);
        sum = fmaf(bf_lo(wv.z), hb.x, sum);
        sum = fmaf(bf_hi(wv.z), hb.y, sum);
        sum = fmaf(bf_lo(wv.w), hb.z, sum);
        sum = fmaf(bf_hi(wv.w), hb.w, sum);
      }
#pragma unroll
      for (int off = 16; off; off >>= 1) sum += __shfl_xor(sum, off, 32);
      if (lane == 0) gsum[gate][rbase + r] = sum;
    }
    __syncthreads();

    // 16 threads apply gate nonlinearities, update c, emit h
    if (tid < 16) {
      const int j = jbase + tid;
      const float* gxt = gx + (size_t)t * G_DIM;
      float gi = gsum[0][tid] + gxt[j];
      float gf = gsum[1][tid] + gxt[H_DIM + j];
      float gg = gsum[2][tid] + gxt[2 * H_DIM + j];
      float go = gsum[3][tid] + gxt[3 * H_DIM + j];
      gi = 1.f / (1.f + __expf(-gi));
      gf = 1.f / (1.f + __expf(-gf));
      gg = tanhf(gg);
      go = 1.f / (1.f + __expf(-go));
      c = gf * c + gi * gg;
      const float hn = go * tanhf(c);
      hout[j] = hn;
      hs[(size_t)t * H_DIM + j] = __float2bfloat16(hn);
    }

    // grid-wide barrier (double-buffered h makes one barrier per step sufficient)
    __threadfence();
    __syncthreads();
    if (tid == 0) {
      atomicAdd(sync, 1u);
      const unsigned target = (unsigned)(t + 1) * NB_REC;
      volatile unsigned* vs = (volatile unsigned*)sync;
      while (*vs < target) { __builtin_amdgcn_s_sleep(1); }
    }
    __syncthreads();
    __threadfence();
  }
}

// ---------------- GEMM3: out[T,O] = hs[T,H] @ Wd[O,H]^T + b_d ----------------
// block = 256 thr (8 waves) covers 16 rows x full 128 cols
__global__ __launch_bounds__(256) void k_gemm_out(
    const __hip_bfloat16* __restrict__ Hs,    // [T, H]
    const __hip_bfloat16* __restrict__ Wd,    // [O, H]
    const float* __restrict__ bd,             // [O]
    float* __restrict__ out)                  // [T, O]
{
  const int lane = threadIdx.x & 31;
  const int wave = threadIdx.x >> 5;
  const int hlf  = lane >> 4;
  const int l15  = lane & 15;
  const int mbase = blockIdx.x * 16;
  const int nbase = wave * 16;

  const __hip_bfloat16* arow = Hs + (size_t)(mbase + l15) * H_DIM;
  const __hip_bfloat16* brow = Wd + (size_t)(nbase + l15) * H_DIM;

  v8f acc = {0.f, 0.f, 0.f, 0.f, 0.f, 0.f, 0.f, 0.f};
  for (int k = 0; k < H_DIM; k += 32) {
    Frag16 a, b;
    a.f[0] = *(const float4*)(arow + k +      hlf * 8);
    a.f[1] = *(const float4*)(arow + k + 16 + hlf * 8);
    b.f[0] = *(const float4*)(brow + k +      hlf * 8);
    b.f[1] = *(const float4*)(brow + k + 16 + hlf * 8);
    acc = __builtin_amdgcn_wmma_f32_16x16x32_bf16(false, a.v, false, b.v,
                                                  (short)0, acc, false, false);
  }
  const float bv = bd[nbase + l15];
#pragma unroll
  for (int r = 0; r < 8; ++r) {
    int m = mbase + r + hlf * 8;
    out[(size_t)m * O_DIM + nbase + l15] = acc[r] + bv;
  }
}

// ---------------- launch ----------------
extern "C" void kernel_launch(void* const* d_in, const int* in_sizes, int n_in,
                              void* d_out, int out_size, void* d_ws, size_t ws_size,
                              hipStream_t stream) {
  const float* x    = (const float*)d_in[0];   // [1,T,I]
  const float* W_ih = (const float*)d_in[1];   // [4H,I]
  const float* W_hh = (const float*)d_in[2];   // [4H,H]
  const float* b_ih = (const float*)d_in[3];   // [4H]
  const float* b_hh = (const float*)d_in[4];   // [4H]
  const float* W_d  = (const float*)d_in[5];   // [O,H]
  const float* b_d  = (const float*)d_in[6];   // [O]
  float* out = (float*)d_out;                  // [1,T,O]

  // workspace carve-up (256B aligned)
  auto align256 = [](size_t v) { return (v + 255) & ~(size_t)255; };
  char* ws = (char*)d_ws;
  size_t off = 0;
  __hip_bfloat16* Wih_b = (__hip_bfloat16*)(ws + off); off = align256(off + (size_t)G_DIM * I_DIM * 2);
  __hip_bfloat16* Whh_b = (__hip_bfloat16*)(ws + off); off = align256(off + (size_t)G_DIM * H_DIM * 2);
  __hip_bfloat16* Wd_b  = (__hip_bfloat16*)(ws + off); off = align256(off + (size_t)O_DIM * H_DIM * 2);
  __hip_bfloat16* X_b   = (__hip_bfloat16*)(ws + off); off = align256(off + (size_t)T_STEPS * I_DIM * 2);
  float*          bias  = (float*)(ws + off);          off = align256(off + (size_t)G_DIM * 4);
  float*          hbuf  = (float*)(ws + off);          off = align256(off + (size_t)2 * H_DIM * 4);
  __hip_bfloat16* Hs_b  = (__hip_bfloat16*)(ws + off); off = align256(off + (size_t)T_STEPS * H_DIM * 2);
  float*          gx    = (float*)(ws + off);          off = align256(off + (size_t)T_STEPS * G_DIM * 4);
  unsigned*       sync  = (unsigned*)(ws + off);       off = align256(off + 256);

  auto blocks = [](int n) { return (n + 255) / 256; };

  // 1) down-convert operands to bf16
  k_cvt_bf16<<<blocks(T_STEPS * I_DIM), 256, 0, stream>>>(x,    X_b,   T_STEPS * I_DIM);
  k_cvt_bf16<<<blocks(G_DIM * I_DIM),   256, 0, stream>>>(W_ih, Wih_b, G_DIM * I_DIM);
  k_cvt_bf16<<<blocks(G_DIM * H_DIM),   256, 0, stream>>>(W_hh, Whh_b, G_DIM * H_DIM);
  k_cvt_bf16<<<blocks(O_DIM * H_DIM),   256, 0, stream>>>(W_d,  Wd_b,  O_DIM * H_DIM);

  // 2) bias sum, zero h double-buffer + sync counter (every call -> deterministic)
  k_setup<<<blocks(G_DIM), 256, 0, stream>>>(b_ih, b_hh, bias, hbuf, sync);

  // 3) gx = x @ W_ih^T + bias   (WMMA bf16)
  dim3 g1(G_DIM / 128, T_STEPS / 16);
  k_gemm_gx<<<g1, 256, 0, stream>>>(X_b, Wih_b, bias, gx);

  // 4) sequential LSTM scan, persistent grid, W_hh streamed from L2
  k_lstm_steps<<<NB_REC, 256, 0, stream>>>(Whh_b, gx, hbuf, Hs_b, sync);

  // 5) out = hs @ W_d^T + b_d   (WMMA bf16)
  k_gemm_out<<<T_STEPS / 16, 256, 0, stream>>>(Hs_b, Wd_b, b_d, out);
}
